// StyleConv_16003048145096
// MI455X (gfx1250) — matrix operand order repaired and checked by reading, not compiled, not run on hardware
//
#include <hip/hip_runtime.h>

typedef __attribute__((ext_vector_type(16))) _Float16 v16h;
typedef __attribute__((ext_vector_type(8)))  float    v8f;

union Frag {
    v16h h;
    uint4 u[2];
};

#define B_   8
#define CIN  512
#define COUT 512
#define HH   64
#define WW   64
#define NTAP 9
#define KCH  32                 // ci per K-chunk
#define NCHUNK (CIN / KCH)
#define LCOL (WW + 2)           // LDS columns incl. zero halo
#define LDS_U4 (3 * LCOL * 4)   // uint4 slots: 792 (12,672 B)

// ---------------------------------------------------------------------------
// Kernel 1: s[b][ci] = style[b,:] . mod_w[ci,:] + mod_b[ci]
// ---------------------------------------------------------------------------
__global__ __launch_bounds__(256) void style_kernel(const float* __restrict__ style,
                                                    const float* __restrict__ mod_w,
                                                    const float* __restrict__ mod_b,
                                                    float* __restrict__ s) {
    int idx = blockIdx.x * 256 + threadIdx.x;     // 0 .. 8*512-1
    int b = idx >> 9;
    int ci = idx & 511;
    const float* st = style + b * 512;
    const float* mw = mod_w + ci * 512;
    float acc = 0.f;
#pragma unroll 8
    for (int j = 0; j < 512; ++j) acc += st[j] * mw[j];
    s[idx] = acc + mod_b[ci];
}

// ---------------------------------------------------------------------------
// Kernel 2: x (B,C,H,W f32) -> xw[b][cc][h][w][ci32] f16   (chunked NHWC)
// one wave per (b, cc, h) row; lane = ci within chunk
// ---------------------------------------------------------------------------
__global__ __launch_bounds__(256) void xcvt_kernel(const float* __restrict__ x,
                                                   _Float16* __restrict__ xw) {
    int wid  = blockIdx.x * 8 + (threadIdx.x >> 5);   // 0 .. 8191
    int lane = threadIdx.x & 31;
    int h  = wid & 63;
    int cc = (wid >> 6) & 15;
    int b  = wid >> 10;
    const float* src = x + (((b * CIN + cc * KCH + lane) * HH + h) * WW);
    _Float16* dst = xw + ((((b * NCHUNK + cc) * HH + h) * WW) * KCH) + lane;
#pragma unroll 4
    for (int w = 0; w < WW; ++w)
        dst[w * KCH] = (_Float16)src[w];
}

// ---------------------------------------------------------------------------
// Kernel 3: modulate + demodulate weights, fold sqrt(2), store f16 as
//           wq[b][tap][co][ci]  (A-matrix friendly: contiguous ci per tap)
// one block per (co, b)
// ---------------------------------------------------------------------------
__global__ __launch_bounds__(256) void modw_kernel(const float* __restrict__ conv_w,
                                                   const float* __restrict__ s,
                                                   _Float16* __restrict__ wq) {
    int co = blockIdx.x;
    int b  = blockIdx.y;
    __shared__ float red[256];
    const float* wrow = conv_w + co * (CIN * NTAP);
    const float* sb   = s + b * CIN;
    float sum = 0.f;
#pragma unroll
    for (int i = 0; i < 18; ++i) {
        int e = threadIdx.x + i * 256;        // e = ci*9 + tap
        float v = wrow[e] * sb[e / 9];
        sum += v * v;
    }
    red[threadIdx.x] = sum;
    __syncthreads();
    for (int off = 128; off > 0; off >>= 1) {
        if (threadIdx.x < off) red[threadIdx.x] += red[threadIdx.x + off];
        __syncthreads();
    }
    float demod = rsqrtf(red[0] + 1e-8f) * 1.41421356237309515f;   // fold sqrt(2)
    __syncthreads();
#pragma unroll
    for (int tap = 0; tap < NTAP; ++tap) {
#pragma unroll
        for (int i = 0; i < 2; ++i) {
            int ci = threadIdx.x + i * 256;
            float v = wrow[ci * NTAP + tap] * sb[ci] * demod;
            wq[(((b * NTAP + tap) * COUT + co) * CIN) + ci] = (_Float16)v;
        }
    }
}

// ---------------------------------------------------------------------------
// Kernel 4: main WMMA implicit-GEMM conv + epilogue
// grid (4, 64, 8): (c_out tile of 128, image row h, batch)
// block 256 = 8 waves; wave m handles c_out rows [128*bx + 16*m, +16), 64 px
// LDS tile xs[3][66][32] f16 with zero halo cols -> branch-free B fragments
// ---------------------------------------------------------------------------
__global__ __launch_bounds__(256) void conv_kernel(const _Float16* __restrict__ xw,
                                                   const _Float16* __restrict__ wq,
                                                   const float* __restrict__ noise,
                                                   const float* __restrict__ noise_w,
                                                   const float* __restrict__ bias,
                                                   float* __restrict__ out) {
    __shared__ uint4 xs4[LDS_U4];

    int ctile = blockIdx.x;
    int h     = blockIdx.y;
    int b     = blockIdx.z;
    int wave   = threadIdx.x >> 5;
    int lane   = threadIdx.x & 31;
    int laneHi = lane >> 4;      // K-half selector (A and B fragments)
    int lanelo = lane & 15;      // M row (A), N column (B/C/D)

    // ---- zero the whole LDS tile once (halo cols + out-of-range rows stay 0)
#pragma unroll
    for (int i = 0; i < 4; ++i) {
        int idx = threadIdx.x + i * 256;
        if (idx < LDS_U4) xs4[idx] = make_uint4(0u, 0u, 0u, 0u);
    }

    v8f zero = {0.f, 0.f, 0.f, 0.f, 0.f, 0.f, 0.f, 0.f};
    v8f acc[4];
#pragma unroll
    for (int nt = 0; nt < 4; ++nt) acc[nt] = zero;

    const uint4* xwg = (const uint4*)xw;
    int co_row = ctile * 128 + wave * 16 + lanelo;
    // A-fragment base: per-tap offset (tap*COUT*CIN halves = 512KB) folds into
    // the 24-bit signed instruction offset of global_load_b128
    const _Float16* wbase =
        wq + ((size_t)b * NTAP * COUT * CIN) + (size_t)co_row * CIN + laneHi * 8;
    // per-lane B-fragment base (uint4 units); (dy,dx,nt) parts are immediates
    const uint4* xsb = xs4 + (lanelo << 2) + laneHi;

    // staging decomposition for this thread (constant across chunks)
    int sq   = threadIdx.x & 3;           // 16B quarter within 64B ci-block
    int scol = (threadIdx.x >> 2) & 63;   // image column

    __syncthreads();

    for (int cc = 0; cc < NCHUNK; ++cc) {
        // ---- stage x rows h-1..h+1 of this ci-chunk into LDS interior
#pragma unroll
        for (int dyi = 0; dyi < 3; ++dyi) {
            int dy = ((threadIdx.x >> 8) + dyi) % 3;   // threadIdx.x >> 8 is 0 (256 threads): dy = dyi
            int rg = h + dy - 1;
            if ((unsigned)rg < (unsigned)HH) {
                uint4 v = xwg[((((b * NCHUNK + cc) * HH + rg) * WW + scol) << 2) + sq];
                xs4[(dy * LCOL + scol + 1) * 4 + sq] = v;
            }
        }
        __syncthreads();

        const _Float16* wp = wbase + cc * KCH;
        if (cc + 1 < NCHUNK)
            __builtin_prefetch(wbase + (cc + 1) * KCH, 0, 0);

#pragma unroll
        for (int dy = 0; dy < 3; ++dy) {
#pragma unroll
            for (int dx = 0; dx < 3; ++dx) {
                const int tap = dy * 3 + dx;
                Frag af;   // A (16x32 MxK), interleaved K split per ISA layout
                af.u[0] = *(const uint4*)(wp + tap * (COUT * CIN));
                af.u[1] = *(const uint4*)(wp + tap * (COUT * CIN) + 16);
#pragma unroll
                for (int nt = 0; nt < 4; ++nt) {
                    // col2 = (input col)+1 in padded tile; always in range
                    const uint4* p = xsb + (dy * LCOL + nt * 16 + dx) * 4;
                    Frag bf;
                    bf.u[0] = p[0];
                    bf.u[1] = p[2];
                    acc[nt] = __builtin_amdgcn_wmma_f32_16x16x32_f16(
                        false, af.h, false, bf.h, (short)0, acc[nt], false, false);
                }
            }
        }
        __syncthreads();
    }

    // ---- epilogue: + noise_w*noise + bias, LeakyReLU(0.2)
    float nw = noise_w[0];
    int c0 = ctile * 128 + wave * 16 + laneHi * 8;
    float bs[8];
#pragma unroll
    for (int r = 0; r < 8; ++r) bs[r] = bias[c0 + r];
#pragma unroll
    for (int nt = 0; nt < 4; ++nt) {
        int wcol = nt * 16 + lanelo;
        float nz = nw * noise[(b * HH + h) * WW + wcol];
#pragma unroll
        for (int r = 0; r < 8; ++r) {
            float v = acc[nt][r] + nz + bs[r];
            v = (v >= 0.f) ? v : 0.2f * v;
            out[((((size_t)b * COUT + (c0 + r)) * HH + h) * WW) + wcol] = v;
        }
    }
}

// ---------------------------------------------------------------------------
extern "C" void kernel_launch(void* const* d_in, const int* in_sizes, int n_in,
                              void* d_out, int out_size, void* d_ws, size_t ws_size,
                              hipStream_t stream) {
    const float* x       = (const float*)d_in[0];
    const float* style   = (const float*)d_in[1];
    const float* noise   = (const float*)d_in[2];
    const float* mod_w   = (const float*)d_in[3];
    const float* mod_b   = (const float*)d_in[4];
    const float* conv_w  = (const float*)d_in[5];
    const float* noise_w = (const float*)d_in[6];
    const float* bias    = (const float*)d_in[7];
    float* out = (float*)d_out;

    char* ws = (char*)d_ws;
    float*    s  = (float*)ws;                                   // 16 KB
    _Float16* wq = (_Float16*)(ws + 16384);                      // 8*9*512*512*2 = 37,748,736 B
    _Float16* xw = (_Float16*)(ws + 16384 + 37748736);           // 8*16*64*64*32*2 = 33,554,432 B

    style_kernel<<<16, 256, 0, stream>>>(style, mod_w, mod_b, s);
    xcvt_kernel<<<1024, 256, 0, stream>>>(x, xw);
    modw_kernel<<<dim3(COUT, B_), 256, 0, stream>>>(conv_w, s, wq);
    conv_kernel<<<dim3(4, HH, B_), 256, 0, stream>>>(xw, wq, noise, noise_w, bias, out);
}
